// DotProductAttention_15367392985385
// MI455X (gfx1250) — matrix operand order repaired and checked by reading, not compiled, not run on hardware
//
#include <hip/hip_runtime.h>

// DotProductAttention fused kernel for gfx1250 (MI455X), wave32 + bf16 WMMA.
// d_out = [out: B*Q*D f32][attn: B*Q*K f32] concatenated.
// d_ws (if >= 64MB): [Kb: B*K*D bf16][Vt: B*D*K bf16] pre-converted operands.

#define B_ 64
#define Q_ 2048
#define K_ 2048
#define D_ 128
#define NEGV (-1e9f)
#define WAVES 8
#define PSTR 33  // padded LDS row stride (bf16 elements)

typedef __attribute__((ext_vector_type(16))) __bf16 v16bf;
typedef __attribute__((ext_vector_type(8)))  __bf16 v8bf;
typedef __attribute__((ext_vector_type(8)))  float  v8f;

// 16-bit A-matrix 16x32 layout (contiguous, matches B): element e of lane-half h -> K index h*16+e
__device__ __forceinline__ constexpr int a_k(int e, int h) {
  return (h << 4) + e;
}

// ---- pre-pass 1: K f32 -> bf16, same row-major layout ----
__global__ __launch_bounds__(256) void cvt_k(const float* __restrict__ src,
                                             __bf16* __restrict__ dst) {
  const size_t i = ((size_t)blockIdx.x * 256 + threadIdx.x) * 8;
  const float4* s = (const float4*)(src + i);
  const float4 x0 = s[0], x1 = s[1];
  v8bf o;
  o[0] = (__bf16)x0.x; o[1] = (__bf16)x0.y; o[2] = (__bf16)x0.z; o[3] = (__bf16)x0.w;
  o[4] = (__bf16)x1.x; o[5] = (__bf16)x1.y; o[6] = (__bf16)x1.z; o[7] = (__bf16)x1.w;
  *(v8bf*)(dst + i) = o;
}

// ---- pre-pass 2: V f32 [b][k][d] -> bf16 transposed [b][d][k] (64x64 LDS tiles) ----
__global__ __launch_bounds__(256) void tr_v(const float* __restrict__ src,
                                            __bf16* __restrict__ dst) {
  __shared__ __bf16 t[64][65];
  const int kt = blockIdx.x & 31;        // K/64
  const int dt = (blockIdx.x >> 5) & 1;  // D/64
  const int b  = blockIdx.x >> 6;
  const int k0 = kt << 6, d0 = dt << 6;
  const int tx = threadIdx.x & 63;
  const int ty = threadIdx.x >> 6;       // 0..3
#pragma unroll
  for (int i = 0; i < 16; ++i) {
    const int kl = i * 4 + ty;
    t[tx][kl] = (__bf16)src[((size_t)b * K_ + k0 + kl) * D_ + d0 + tx];
  }
  __syncthreads();
#pragma unroll
  for (int i = 0; i < 16; ++i) {
    const int dl = i * 4 + ty;
    dst[((size_t)b * D_ + d0 + dl) * K_ + k0 + tx] = t[dl][tx];
  }
}

template <bool PRE>
__global__ __launch_bounds__(256)
void attn_fused(const float* __restrict__ Qg, const float* __restrict__ Kg,
                const float* __restrict__ Vg, const __bf16* __restrict__ Kb,
                const __bf16* __restrict__ Vt, const int* __restrict__ VL,
                float* __restrict__ Og, float* __restrict__ Ag) {
  __shared__ __bf16 plds_all[WAVES * 16 * PSTR];

  const int lane = threadIdx.x & 31;
  const int wave = threadIdx.x >> 5;
  const int lr   = lane & 15;   // row (A) / col (B,C,D) within half
  const int h    = lane >> 4;   // lane half
  const int tile = blockIdx.x * WAVES + wave;
  const int b    = tile >> 7;            // 128 q-tiles per batch
  const int q0   = (tile & 127) << 4;    // *16
  const int vlen = VL[b];
  const float scale = 0.08838834764831845f;  // 1/sqrt(128)

  __bf16* pl = plds_all + wave * 16 * PSTR;

  // ---- Q tile -> bf16 A fragments, scale folded in (A row m = lr) ----
  v16bf aq[4];
  {
    const float* qrow = Qg + ((size_t)b * Q_ + q0 + lr) * D_;
#pragma unroll
    for (int c = 0; c < 4; ++c) {
#pragma unroll
      for (int e = 0; e < 16; ++e)
        aq[c][e] = (__bf16)(qrow[c * 32 + a_k(e, h)] * scale);
    }
  }

  // ---- pass 1: masked scores -> per-lane online (max, sum) ----
  float mrun[8], srun[8];
#pragma unroll
  for (int r = 0; r < 8; ++r) { mrun[r] = -1e30f; srun[r] = 0.0f; }

  for (int k0 = 0; k0 < K_; k0 += 16) {
    if (PRE && (k0 + 64) < K_)  // warm the cache for the tile 4 steps ahead
      __builtin_prefetch(Kb + ((size_t)b * K_ + k0 + 64 + lr) * D_, 0, 3);
    v8f s = {0.f, 0.f, 0.f, 0.f, 0.f, 0.f, 0.f, 0.f};
#pragma unroll
    for (int c = 0; c < 4; ++c) {
      v16bf bk;   // B col n = lr (key row), B row kk = h*16+e (d index)
      if (PRE) {
        bk = *(const v16bf*)(Kb + ((size_t)b * K_ + k0 + lr) * D_ + c * 32 + (h << 4));
      } else {
        const float* krow = Kg + ((size_t)b * K_ + k0 + lr) * D_;
#pragma unroll
        for (int e = 0; e < 16; ++e)
          bk[e] = (__bf16)krow[c * 32 + (h << 4) + e];
      }
      s = __builtin_amdgcn_wmma_f32_16x16x32_bf16(false, aq[c], false, bk,
                                                  (short)0, s, false, false);
    }
    const bool oob = (k0 + lr) >= vlen;            // C/D col n = lr
#pragma unroll
    for (int r = 0; r < 8; ++r) {                  // C/D row m = r + 8*h
      const float x  = oob ? NEGV : s[r];
      const float mo = mrun[r];
      const float mn = fmaxf(mo, x);
      srun[r] = srun[r] * __expf(mo - mn) + __expf(x - mn);
      mrun[r] = mn;
    }
  }
  // combine the 16 column-partials (they live in lanes of the same half)
#pragma unroll
  for (int off = 1; off < 16; off <<= 1) {
#pragma unroll
    for (int r = 0; r < 8; ++r) {
      const float mo = __shfl_xor(mrun[r], off, 32);
      const float so = __shfl_xor(srun[r], off, 32);
      const float mn = fmaxf(mrun[r], mo);
      srun[r] = srun[r] * __expf(mrun[r] - mn) + so * __expf(mo - mn);
      mrun[r] = mn;
    }
  }
  float inv[8];
#pragma unroll
  for (int r = 0; r < 8; ++r) inv[r] = 1.0f / srun[r];

  // ---- pass 2: normalized P -> attn store (NT) + out += P*V ----
  v8f oacc[8];
#pragma unroll
  for (int dt = 0; dt < 8; ++dt) oacc[dt] = (v8f){0.f,0.f,0.f,0.f,0.f,0.f,0.f,0.f};

  float* arow = Ag + ((size_t)b * Q_ + q0) * K_;

  for (int k0 = 0; k0 < K_; k0 += 32) {
#pragma unroll
    for (int t = 0; t < 2; ++t) {
      const int kk0 = k0 + t * 16;
      v8f s = {0.f, 0.f, 0.f, 0.f, 0.f, 0.f, 0.f, 0.f};
#pragma unroll
      for (int c = 0; c < 4; ++c) {
        v16bf bk;
        if (PRE) {
          bk = *(const v16bf*)(Kb + ((size_t)b * K_ + kk0 + lr) * D_ + c * 32 + (h << 4));
        } else {
          const float* krow = Kg + ((size_t)b * K_ + kk0 + lr) * D_;
#pragma unroll
          for (int e = 0; e < 16; ++e)
            bk[e] = (__bf16)krow[c * 32 + (h << 4) + e];
        }
        s = __builtin_amdgcn_wmma_f32_16x16x32_bf16(false, aq[c], false, bk,
                                                    (short)0, s, false, false);
      }
      const bool oob = (kk0 + lr) >= vlen;
#pragma unroll
      for (int r = 0; r < 8; ++r) {
        const float x = oob ? NEGV : s[r];
        const float p = __expf(x - mrun[r]) * inv[r];
        const int   m = r + (h << 3);
        __builtin_nontemporal_store(p, &arow[(size_t)m * K_ + kk0 + lr]); // streaming 1GiB
        pl[m * PSTR + t * 16 + lr] = (__bf16)p;        // stage for C->A transpose
      }
    }
    asm volatile("s_wait_dscnt 0x0" ::: "memory");
    __builtin_amdgcn_wave_barrier();
    // reload P in A layout
    v16bf ap;
#pragma unroll
    for (int e = 0; e < 16; ++e)
      ap[e] = pl[lr * PSTR + a_k(e, h)];
    __builtin_amdgcn_wave_barrier();
    // V as B tiles (B row kk = h*16+e -> key row k0+kk; col n = lr -> d)
#pragma unroll
    for (int dt = 0; dt < 8; ++dt) {
      v16bf bv;
      if (PRE) {
        bv = *(const v16bf*)(Vt + ((size_t)b * D_ + dt * 16 + lr) * K_ + k0 + (h << 4));
      } else {
#pragma unroll
        for (int e = 0; e < 16; ++e)
          bv[e] = (__bf16)Vg[((size_t)b * K_ + k0 + (h << 4) + e) * D_ + dt * 16 + lr];
      }
      oacc[dt] = __builtin_amdgcn_wmma_f32_16x16x32_bf16(false, ap, false, bv,
                                                         (short)0, oacc[dt], false, false);
    }
    __builtin_amdgcn_wave_barrier();
  }

  // ---- store out (C/D layout: col = lr, row = r + 8*h) ----
#pragma unroll
  for (int dt = 0; dt < 8; ++dt) {
#pragma unroll
    for (int r = 0; r < 8; ++r) {
      const int m = r + (h << 3);
      __builtin_nontemporal_store(oacc[dt][r],
          &Og[((size_t)b * Q_ + q0 + m) * D_ + dt * 16 + lr]);
    }
  }
}

extern "C" void kernel_launch(void* const* d_in, const int* in_sizes, int n_in,
                              void* d_out, int out_size, void* d_ws, size_t ws_size,
                              hipStream_t stream) {
  const float* Qg = (const float*)d_in[0];
  const float* Kg = (const float*)d_in[1];
  const float* Vg = (const float*)d_in[2];
  const int*   VL = (const int*)d_in[3];
  float* Og = (float*)d_out;
  float* Ag = Og + (size_t)B_ * Q_ * D_;   // attn follows out

  const size_t elems = (size_t)B_ * K_ * D_;       // 16.7M per operand
  const size_t need  = elems * 2 * sizeof(__bf16); // Kb + Vt = 64MB
  const dim3 block(WAVES * 32);
  const dim3 grid(B_ * (Q_ / 16) / WAVES);         // 1024 blocks, 8 q-tiles each

  if (ws_size >= need) {
    __bf16* Kb = (__bf16*)d_ws;
    __bf16* Vt = Kb + elems;
    cvt_k<<<dim3((unsigned)(elems / (256 * 8))), dim3(256), 0, stream>>>(Kg, Kb);
    tr_v<<<dim3(B_ * (D_ / 64) * (K_ / 64)), dim3(256), 0, stream>>>(Vg, Vt);
    attn_fused<true><<<grid, block, 0, stream>>>(Qg, Kg, Vg, Kb, Vt, VL, Og, Ag);
  } else {
    attn_fused<false><<<grid, block, 0, stream>>>(Qg, Kg, Vg, nullptr, nullptr, VL, Og, Ag);
  }

  (void)in_sizes; (void)n_in; (void)out_size;
}
